// KoopmanDynamicsPredictor_59227599012588
// MI455X (gfx1250) — compile-verified, matrix operand order checked by reading, not run on hardware
//
#include <hip/hip_runtime.h>
#include <hip/hip_fp16.h>
#include <math.h>

// ---------------------------------------------------------------------------
// Koopman dynamics predictor for MI455X (gfx1250, wave32, WMMA).
// B=256, H=512, STATE=64, ACT=16, KOOP=448, LATENT=512, HID=1024.
// Scan decomposed into chunked two-pass form (C=16) for full-machine WMMA use.
// All matrix math on v_wmma_f32_16x16x32_f16 with f32 accumulators; B-side
// matrices (At powers) kept as f16 hi+lo pairs (2-product) for fp32-like A.
// ---------------------------------------------------------------------------

typedef _Float16 f16;
typedef __attribute__((ext_vector_type(8)))  _Float16 v8h;
typedef __attribute__((ext_vector_type(16))) _Float16 v16h;
typedef __attribute__((ext_vector_type(8)))  float    v8f;

#define WMMAF16(A, Bm, Cm) \
  __builtin_amdgcn_wmma_f32_16x16x32_f16(false, (A), false, (Bm), (short)0, (Cm), false, false)

static __device__ __forceinline__ v16h cat16(v8h a, v8h b) {
  return __builtin_shufflevector(a, b, 0,1,2,3,4,5,6,7,8,9,10,11,12,13,14,15);
}
static __device__ __forceinline__ v8h ldv8(const f16* p) { return *(const v8h*)p; }

#define BB     256
#define HH     512
#define LAT    512
#define HID    1024
#define CH     16          // chunk length
#define NCH    32          // number of chunks (HH / CH)
#define SMAT   (512*512)   // one latent matrix
#define SCAR   (256*512)   // one carry (B x LATENT)

// ------------------------------ converters ---------------------------------

__global__ void k_cvt_f16(const float* __restrict__ s, f16* __restrict__ d, int n) {
  int i = blockIdx.x * 256 + threadIdx.x;
  if (i < n) d[i] = (f16)s[i];
}

// A_w (row-major [n][k]) is exactly the "T layout" of At (= A_w^T).
// Build T_1 (hi/lo) directly and P_1 (row-major At[k][n]) by transpose.
__global__ void k_cvtA(const float* __restrict__ Aw,
                       f16* __restrict__ TH, f16* __restrict__ TL,
                       f16* __restrict__ PH, f16* __restrict__ PL) {
  int i = blockIdx.x * 256 + threadIdx.x;      // 0..262143
  int n = i >> 9, k = i & 511;
  float v = Aw[i];
  f16 hi = (f16)v;
  f16 lo = (f16)(v - (float)hi);
  TH[i] = hi; TL[i] = lo;
  PH[k * 512 + n] = hi; PL[k * 512 + n] = lo;
}

// carry slot 0 = z0 = concat(x, phi), rounded to f16
__global__ void k_z0(const float* __restrict__ state, const float* __restrict__ phi,
                     f16* __restrict__ c0) {
  int i = blockIdx.x * 256 + threadIdx.x;      // 0..131071
  int b = i >> 9, l = i & 511;
  float v = (l < 64) ? state[b * 64 + l] : phi[b * 448 + (l - 64)];
  c0[i] = (f16)v;
}

// ------------------------------ MLP layers ---------------------------------
// out = act(X[256,K] @ W[N,K]^T + bias).  grid(ceil(N/128),16), block 256.
__global__ __launch_bounds__(256) void k_mlp(const f16* __restrict__ X,
                                             const f16* __restrict__ W,
                                             const float* __restrict__ bias,
                                             f16* __restrict__ out16,
                                             float* __restrict__ out32,
                                             int N, int K, int gelu) {
  int lane = threadIdx.x & 31, wv = threadIdx.x >> 5;
  int nt = blockIdx.x * 8 + wv;
  if (nt * 16 >= N) return;
  int mt = blockIdx.y;
  int nl = lane & 15, hh = lane >> 4;
  v8f acc = {};
  int ar = (mt * 16 + nl) * K;
  int br = (nt * 16 + nl) * K;
  for (int k0 = 0; k0 < K; k0 += 32) {
    v16h a = cat16(ldv8(X + ar + k0 + 8 * hh), ldv8(X + ar + k0 + 16 + 8 * hh));
    v16h b = cat16(ldv8(W + br + k0 + 8 * hh), ldv8(W + br + k0 + 16 + 8 * hh));
    acc = WMMAF16(a, b, acc);
  }
  float bs = bias[nt * 16 + nl];
  for (int r = 0; r < 8; r++) {
    float v = acc[r] + bs;
    if (gelu) v = 0.5f * v * (1.0f + erff(v * 0.70710678118654752f));
    int m = mt * 16 + 8 * hh + r, col = nt * 16 + nl;
    if (out16) out16[m * N + col] = (f16)v;
    else       out32[m * N + col] = v;
  }
}

// ------------------------- power doubling (At^j) ---------------------------
// P_{a+b} = P_a @ At^b  (B-operand streamed from T layout, hi/lo 3-product).
// grid(4, 32, jcount), block 256.  a is the level base; b = 1 + blockIdx.z.
__global__ __launch_bounds__(256) void k_pow(f16* __restrict__ PH, f16* __restrict__ PL,
                                             f16* __restrict__ TH, f16* __restrict__ TL,
                                             int a) {
  int bidx = 1 + blockIdx.z;
  int j = a + bidx;
  const f16* pha = PH + (size_t)(a - 1) * SMAT;
  const f16* pla = PL + (size_t)(a - 1) * SMAT;
  const f16* thb = TH + (size_t)(bidx - 1) * SMAT;
  const f16* tlb = TL + (size_t)(bidx - 1) * SMAT;
  f16* phd = PH + (size_t)(j - 1) * SMAT;
  f16* pld = PL + (size_t)(j - 1) * SMAT;
  f16* thd = TH + (size_t)(j - 1) * SMAT;
  f16* tld = TL + (size_t)(j - 1) * SMAT;

  int lane = threadIdx.x & 31, wv = threadIdx.x >> 5;
  int nt = blockIdx.x * 8 + wv, mt = blockIdx.y;
  int nl = lane & 15, hh = lane >> 4;
  int ar = (mt * 16 + nl) * 512;
  int br = (nt * 16 + nl) * 512;
  v8f acc = {};
  for (int k0 = 0; k0 < 512; k0 += 32) {
    v16h ah = cat16(ldv8(pha + ar + k0 + 8 * hh), ldv8(pha + ar + k0 + 16 + 8 * hh));
    v16h al = cat16(ldv8(pla + ar + k0 + 8 * hh), ldv8(pla + ar + k0 + 16 + 8 * hh));
    v16h bh = cat16(ldv8(thb + br + k0 + 8 * hh), ldv8(thb + br + k0 + 16 + 8 * hh));
    v16h bl = cat16(ldv8(tlb + br + k0 + 8 * hh), ldv8(tlb + br + k0 + 16 + 8 * hh));
    acc = WMMAF16(ah, bh, acc);
    acc = WMMAF16(al, bh, acc);
    acc = WMMAF16(ah, bl, acc);
  }
  for (int r = 0; r < 8; r++) {
    int k = mt * 16 + 8 * hh + r, n = nt * 16 + nl;
    float v = acc[r];
    f16 hi = (f16)v;
    f16 lo = (f16)(v - (float)hi);
    phd[k * 512 + n] = hi; pld[k * 512 + n] = lo;
    thd[n * 512 + k] = hi; tld[n * 512 + k] = lo;
  }
}

// --------------------------- phase 2: local scans --------------------------
// grid(NCH, 8), block 1024 (32 waves; wave = ntile; each wave covers 2 mtiles).
// s_t = s_{t-1} @ At + Bu_t, s_{t0} = Bu_{t0};  s exchanged through LDS.
__global__ __launch_bounds__(1024) void k_local(const f16* __restrict__ act,
                                                const f16* __restrict__ bw,
                                                const f16* __restrict__ TH1,
                                                const f16* __restrict__ TL1,
                                                float* __restrict__ out) {
  __shared__ f16 stag[32][528];               // 32 rows x 512 latent (+pad)
  int lane = threadIdx.x & 31, nt = threadIdx.x >> 5;
  int nl = lane & 15, hh = lane >> 4;
  int c = blockIdx.x;                          // chunk
  int mg = blockIdx.y;                         // rows mg*32 .. mg*32+31
  int brow = (nt * 16 + nl) * 512;

  // resident B_w operand (real K = 16, upper 16 K zero-padded)
  v8h z8 = {};
  v16h bwop = cat16(ldv8(bw + (nt * 16 + nl) * 16 + 8 * hh), z8);

  int t0 = c * CH;
  v8f acc[2];
  for (int t = t0; t < t0 + CH; t++) {
    // Bu contribution (fresh accumulator each step)
    for (int mi = 0; mi < 2; mi++) {
      int b = mg * 32 + mi * 16 + nl;
      v16h aop = cat16(ldv8(act + ((size_t)b * HH + t) * 16 + 8 * hh), z8);
      v8f zc = {};
      acc[mi] = WMMAF16(aop, bwop, zc);
    }
    if (t + 1 < t0 + CH)
      __builtin_prefetch(act + ((size_t)(mg * 32 + nl) * HH + t + 1) * 16, 0, 1);
    if (t > t0) {
      for (int k0 = 0; k0 < 512; k0 += 32) {
        v16h bh = cat16(ldv8(TH1 + brow + k0 + 8 * hh), ldv8(TH1 + brow + k0 + 16 + 8 * hh));
        v16h bl = cat16(ldv8(TL1 + brow + k0 + 8 * hh), ldv8(TL1 + brow + k0 + 16 + 8 * hh));
        for (int mi = 0; mi < 2; mi++) {
          const f16* sr = &stag[mi * 16 + nl][0];
          v16h a = cat16(ldv8(sr + k0 + 8 * hh), ldv8(sr + k0 + 16 + 8 * hh));
          acc[mi] = WMMAF16(a, bh, acc[mi]);
          acc[mi] = WMMAF16(a, bl, acc[mi]);
        }
      }
      __syncthreads();                         // stag reads done before rewrite
    }
    for (int mi = 0; mi < 2; mi++)
      for (int r = 0; r < 8; r++) {
        int ml = mi * 16 + 8 * hh + r;
        int b = mg * 32 + ml, col = nt * 16 + nl;
        out[((size_t)b * HH + t) * LAT + col] = acc[mi][r];
        stag[ml][col] = (f16)acc[mi][r];
      }
    __syncthreads();                           // stag writes visible
  }
}

// ------------------------ phase 3a: carry recurrence -----------------------
// carry_{c+1} = carry_c @ At^16 + s_end_c.  grid(16), block 1024.
__global__ __launch_bounds__(1024) void k_carry(f16* __restrict__ carry,
                                                const f16* __restrict__ TH,
                                                const f16* __restrict__ TL,
                                                const float* __restrict__ out) {
  int lane = threadIdx.x & 31, nt = threadIdx.x >> 5;
  int nl = lane & 15, hh = lane >> 4;
  int mt = blockIdx.x;                         // rows mt*16 .. mt*16+15
  const f16* thc = TH + (size_t)(CH - 1) * SMAT;   // T_16
  const f16* tlc = TL + (size_t)(CH - 1) * SMAT;
  int brow = (nt * 16 + nl) * 512;
  int ar = (mt * 16 + nl) * 512;
  for (int c = 0; c < NCH - 1; c++) {
    const f16* src = carry + (size_t)c * SCAR;
    f16* dst = carry + (size_t)(c + 1) * SCAR;
    v8f acc = {};
    for (int k0 = 0; k0 < 512; k0 += 32) {
      v16h bh = cat16(ldv8(thc + brow + k0 + 8 * hh), ldv8(thc + brow + k0 + 16 + 8 * hh));
      v16h bl = cat16(ldv8(tlc + brow + k0 + 8 * hh), ldv8(tlc + brow + k0 + 16 + 8 * hh));
      v16h a  = cat16(ldv8(src + ar + k0 + 8 * hh), ldv8(src + ar + k0 + 16 + 8 * hh));
      acc = WMMAF16(a, bh, acc);
      acc = WMMAF16(a, bl, acc);
    }
    int te = c * CH + CH - 1;
    for (int r = 0; r < 8; r++) {
      int b = mt * 16 + 8 * hh + r, l = nt * 16 + nl;
      float v = acc[r] + out[((size_t)b * HH + te) * LAT + l];
      dst[b * 512 + l] = (f16)v;
    }
    __threadfence();
    __syncthreads();
  }
}

// ------------------------- phase 3b: corrections ---------------------------
// z_t += carry_{t/16} @ At^{(t%16)+1}.  grid(512, 2), block 1024.
__global__ __launch_bounds__(1024) void k_corr(const f16* __restrict__ carry,
                                               const f16* __restrict__ TH,
                                               const f16* __restrict__ TL,
                                               float* __restrict__ out) {
  int t = blockIdx.x, mh = blockIdx.y;         // mh: mtiles mh*8 .. mh*8+7
  int c = t >> 4, j = t & 15;
  const f16* thp = TH + (size_t)j * SMAT;      // T_{j+1}
  const f16* tlp = TL + (size_t)j * SMAT;
  const f16* cc = carry + (size_t)c * SCAR;
  int lane = threadIdx.x & 31, nt = threadIdx.x >> 5;
  int nl = lane & 15, hh = lane >> 4;
  int brow = (nt * 16 + nl) * 512;
  v8f acc[8];
  for (int mi = 0; mi < 8; mi++) acc[mi] = (v8f){};
  for (int k0 = 0; k0 < 512; k0 += 32) {
    v16h bh = cat16(ldv8(thp + brow + k0 + 8 * hh), ldv8(thp + brow + k0 + 16 + 8 * hh));
    v16h bl = cat16(ldv8(tlp + brow + k0 + 8 * hh), ldv8(tlp + brow + k0 + 16 + 8 * hh));
    for (int mi = 0; mi < 8; mi++) {
      int m = (mh * 8 + mi) * 16 + nl;
      v16h a = cat16(ldv8(cc + m * 512 + k0 + 8 * hh), ldv8(cc + m * 512 + k0 + 16 + 8 * hh));
      acc[mi] = WMMAF16(a, bh, acc[mi]);
      acc[mi] = WMMAF16(a, bl, acc[mi]);
    }
  }
  for (int mi = 0; mi < 8; mi++)
    for (int r = 0; r < 8; r++) {
      int b = (mh * 8 + mi) * 16 + 8 * hh + r, l = nt * 16 + nl;
      size_t idx = ((size_t)b * HH + t) * LAT + l;
      out[idx] += acc[mi][r];
    }
}

// ------------------------------- launcher ----------------------------------

extern "C" void kernel_launch(void* const* d_in, const int* in_sizes, int n_in,
                              void* d_out, int out_size, void* d_ws, size_t ws_size,
                              hipStream_t stream) {
  const float* state = (const float*)d_in[0];
  const float* action = (const float*)d_in[1];
  const float* W0 = (const float*)d_in[2];  const float* b0 = (const float*)d_in[3];
  const float* W1 = (const float*)d_in[4];  const float* b1 = (const float*)d_in[5];
  const float* W2 = (const float*)d_in[6];  const float* b2 = (const float*)d_in[7];
  const float* W3 = (const float*)d_in[8];  const float* b3 = (const float*)d_in[9];
  const float* Wo = (const float*)d_in[10]; const float* bo = (const float*)d_in[11];
  const float* Aw = (const float*)d_in[12]; const float* Bw = (const float*)d_in[13];
  float* out = (float*)d_out;

  char* ws = (char*)d_ws;
  size_t cur = 0;
  auto alloc = [&](size_t bytes) -> char* {
    char* p = ws + cur;
    cur = (cur + bytes + 255) & ~(size_t)255;
    return p;
  };
  f16* w0f = (f16*)alloc((size_t)HID * 64 * 2);
  f16* w1f = (f16*)alloc((size_t)HID * HID * 2);
  f16* w2f = (f16*)alloc((size_t)HID * HID * 2);
  f16* w3f = (f16*)alloc((size_t)HID * HID * 2);
  f16* wof = (f16*)alloc((size_t)448 * HID * 2);
  f16* xf  = (f16*)alloc((size_t)BB * 64 * 2);
  f16* h1  = (f16*)alloc((size_t)BB * HID * 2);
  f16* h2  = (f16*)alloc((size_t)BB * HID * 2);
  float* phi = (float*)alloc((size_t)BB * 448 * 4);
  f16* actf = (f16*)alloc((size_t)BB * HH * 16 * 2);
  f16* bwf  = (f16*)alloc((size_t)LAT * 16 * 2);
  f16* TH = (f16*)alloc((size_t)CH * SMAT * 2);      // T_1..T_16 hi
  f16* TL = (f16*)alloc((size_t)CH * SMAT * 2);      // T_1..T_16 lo
  f16* PH = (f16*)alloc((size_t)CH * SMAT * 2);      // P_1..P_16 hi (row-major)
  f16* PL = (f16*)alloc((size_t)CH * SMAT * 2);      // P_1..P_16 lo
  f16* carry = (f16*)alloc((size_t)NCH * SCAR * 2);  // carry_0..carry_31
  (void)ws_size; (void)in_sizes; (void)n_in; (void)out_size;

  auto cvt = [&](const float* s, f16* d, int n) {
    k_cvt_f16<<<(n + 255) / 256, 256, 0, stream>>>(s, d, n);
  };
  // converts
  cvt(W0, w0f, HID * 64);
  cvt(W1, w1f, HID * HID);
  cvt(W2, w2f, HID * HID);
  cvt(W3, w3f, HID * HID);
  cvt(Wo, wof, 448 * HID);
  cvt(state, xf, BB * 64);
  cvt(action, actf, BB * HH * 16);
  cvt(Bw, bwf, LAT * 16);
  k_cvtA<<<SMAT / 256, 256, 0, stream>>>(Aw, TH, TL, PH, PL);

  // MLP: x -> h1 -> h2 -> h1 -> h2 -> phi
  k_mlp<<<dim3(8, 16), 256, 0, stream>>>(xf, w0f, b0, h1, nullptr, HID, 64, 1);
  k_mlp<<<dim3(8, 16), 256, 0, stream>>>(h1, w1f, b1, h2, nullptr, HID, HID, 1);
  k_mlp<<<dim3(8, 16), 256, 0, stream>>>(h2, w2f, b2, h1, nullptr, HID, HID, 1);
  k_mlp<<<dim3(8, 16), 256, 0, stream>>>(h1, w3f, b3, h2, nullptr, HID, HID, 1);
  k_mlp<<<dim3(4, 16), 256, 0, stream>>>(h2, wof, bo, nullptr, phi, 448, HID, 0);

  // carry slot 0 = z0
  k_z0<<<SCAR / 256, 256, 0, stream>>>(state, phi, carry);

  // powers by doubling: P_{a+b} = P_a @ At^b for b = 1..a
  for (int a = 1; a < CH; a <<= 1)
    k_pow<<<dim3(4, 32, a), 256, 0, stream>>>(PH, PL, TH, TL, a);

  // phase 2: chunk-local scans (writes all z_t to d_out)
  k_local<<<dim3(NCH, 8), 1024, 0, stream>>>(actf, bwf, TH, TL, out);

  // phase 3a: carry recurrence across chunks
  k_carry<<<dim3(16), 1024, 0, stream>>>(carry, TH, TL, out);

  // phase 3b: parallel corrections
  k_corr<<<dim3(HH, 2), 1024, 0, stream>>>(carry, TH, TL, out);
}
